// AdvancedAttn_63866163692210
// MI455X (gfx1250) — matrix operand order repaired and checked by reading, not compiled, not run on hardware
//
#include <hip/hip_runtime.h>

typedef __bf16 bf16;
typedef __attribute__((ext_vector_type(16))) __bf16 v16bf;
typedef __attribute__((ext_vector_type(8)))  float  v8f;
typedef __attribute__((ext_vector_type(4)))  unsigned int v4u;
typedef __attribute__((ext_vector_type(8)))  int v8i;
typedef __attribute__((ext_vector_type(4)))  int v4i;

#define B_   2
#define S_   2048
#define D_   2048
#define H_   16
#define KVH_ 4
#define DH_  128

static __device__ __forceinline__ v8f zero8(){
  v8f z;
#pragma unroll
  for (int i=0;i<8;++i) z[i]=0.f;
  return z;
}

union FragU { v16bf v; int4 q[2]; };

static __device__ __forceinline__ v16bf load_frag2(const void* p0, const void* p1){
  FragU f; f.q[0] = *(const int4*)p0; f.q[1] = *(const int4*)p1; return f.v;
}
static __device__ __forceinline__ v16bf load_frag32(const void* p){
  FragU f; f.q[0] = ((const int4*)p)[0]; f.q[1] = ((const int4*)p)[1]; return f.v;
}

#define WMMA_BF16(a,b,c) __builtin_amdgcn_wmma_f32_16x16x32_bf16(false,(a),false,(b),(short)0,(c),false,false)

// low 32 bits of a flat pointer into LDS == LDS byte offset (aperture in high bits)
static __device__ __forceinline__ unsigned lds_off32(const void* p){
  return (unsigned)(unsigned long long)(uintptr_t)p;
}
// one 16-byte async global->LDS copy per lane (tracked by ASYNCcnt)
static __device__ __forceinline__ void async_ld16(unsigned lds_off, const void* g){
  asm volatile("global_load_async_to_lds_b128 %0, %1, off"
               :: "v"(lds_off), "v"(g) : "memory");
}

// ------------------------------------------------------------------
// elementwise f32 -> bf16
__global__ void cvt_f32_to_bf16_kernel(const float* __restrict__ in, bf16* __restrict__ out, int n){
  int i = blockIdx.x*blockDim.x + threadIdx.x;
  if (i < n) out[i] = (bf16)in[i];
}

// Wt[n*K + k] = (bf16)W[k*N + n]   (store weights N-major so B-frags are contiguous)
__global__ void transpose_to_bf16_kernel(const float* __restrict__ W, bf16* __restrict__ Wt, int K, int N){
  int gid = blockIdx.x*blockDim.x + threadIdx.x;
  if (gid >= K*N) return;
  int n = gid / K, k = gid - n*K;
  Wt[gid] = (bf16)W[(size_t)k*N + n];
}

// ------------------------------------------------------------------
// Fused QKV projection + RoPE epilogue.
// grid = (S/128, H + 2*KVH, B), block = 256 (8 waves, each wave: 16 rows x one 128-col head)
__global__ void __launch_bounds__(256) proj_rope_kernel(
    const bf16* __restrict__ xb,
    const bf16* __restrict__ WqT, const bf16* __restrict__ WkT, const bf16* __restrict__ WvT,
    bf16* __restrict__ Qp, bf16* __restrict__ Kp, bf16* __restrict__ Vt)
{
  const int lane = threadIdx.x & 31;
  const int w    = threadIdx.x >> 5;
  const int hidx = blockIdx.y;
  const int b    = blockIdx.z;
  const int s0   = blockIdx.x*128 + w*16;
  const int c    = lane & 15;
  const int half = lane >> 4;
  const int khalf= half*8;

  const bf16* Wt; int nbase; int mode;
  if (hidx < H_)           { Wt = WqT; nbase = hidx*DH_;            mode = 0; }
  else if (hidx < H_+KVH_) { Wt = WkT; nbase = (hidx-H_)*DH_;       mode = 1; }
  else                     { Wt = WvT; nbase = (hidx-H_-KVH_)*DH_;  mode = 2; }

  const bf16* arow = xb + (size_t)(b*S_ + s0 + c)*D_;

  v8f acc[8];
#pragma unroll
  for (int t=0;t<8;++t) acc[t] = zero8();

  for (int k0=0;k0<D_;k0+=32){
    v16bf a = load_frag32(arow + k0 + 16*half);
    __builtin_prefetch(arow + k0 + 128, 0, 0);
#pragma unroll
    for (int t=0;t<8;++t){
      const bf16* bp = Wt + (size_t)(nbase + 16*t + c)*D_ + k0 + 16*half;
      v16bf bv = load_frag32(bp);
      acc[t] = WMMA_BF16(a, bv, acc[t]);
    }
  }

  const int rbase = half*8;
  float res[8][8];
  if (mode != 2){
    // RoPE: d = 16t+c; rotate-half partner lives in accumulator tile t^4
#pragma unroll
    for (int t=0;t<8;++t){
      const int d = 16*t + c;
      const float fi = (float)(d & 63);
      const float inv = __powf(10000.f, -fi*(1.f/64.f));
      const float sign = (t < 4) ? -1.f : 1.f;
#pragma unroll
      for (int r=0;r<8;++r){
        const int spos = s0 + rbase + r;
        const float ang = (float)spos * inv;
        res[t][r] = acc[t][r]*__cosf(ang) + sign*acc[t^4][r]*__sinf(ang);
      }
    }
  } else {
#pragma unroll
    for (int t=0;t<8;++t)
#pragma unroll
      for (int r=0;r<8;++r) res[t][r] = acc[t][r];
  }

#pragma unroll
  for (int t=0;t<8;++t){
#pragma unroll
    for (int r=0;r<8;++r){
      const int spos = s0 + rbase + r;
      const int d    = 16*t + c;
      const bf16 v   = (bf16)res[t][r];
      if (mode == 0)
        Qp[((size_t)(b*H_ + hidx)*S_ + spos)*DH_ + d] = v;                      // [B][H][S][DH]
      else if (mode == 1)
        Kp[((size_t)(b*KVH_ + (hidx-H_))*S_ + spos)*DH_ + d] = v;               // [B][KVH][S][DH]
      else
        Vt[((size_t)(b*KVH_ + (hidx-H_-KVH_))*DH_ + d)*S_ + spos] = v;          // [B][KVH][DH][S] (transposed)
    }
  }
}

// ------------------------------------------------------------------
// Flash attention. Q tile staged via TDM (tensor_load_to_lds, TENSORcnt);
// K/V tiles double-buffered via global_load_async_to_lds_b128 (ASYNCcnt).
// grid = (S/128, H, B), block = 256; each wave owns 16 query rows.
__global__ void __launch_bounds__(256) flash_attn_kernel(
    const bf16* __restrict__ Qp, const bf16* __restrict__ Kp, const bf16* __restrict__ Vt,
    bf16* __restrict__ AO)
{
  __shared__ __align__(16) bf16 bufQ[128*DH_];      // 32 KB
  __shared__ __align__(16) bf16 bufK[2][32*DH_];    // 2 x 8 KB
  __shared__ __align__(16) bf16 bufV[2][DH_*32];    // 2 x 8 KB
  __shared__ __align__(16) bf16 ldsP[8][16][32];    // 8 KB P-transpose bounce

  const int tid  = threadIdx.x;
  const int lane = tid & 31;
  const int w    = tid >> 5;
  const int q0   = blockIdx.x*128;
  const int h    = blockIdx.y;
  const int b    = blockIdx.z;
  const int kv   = h >> 2;           // nrep = H/KVH = 4
  const int c    = lane & 15;
  const int half = lane >> 4;
  const int khalf= half*8;
  const int rbase= half*8;

  const bf16* qbase = Qp + ((size_t)(b*H_ + h)*S_ + q0)*DH_;     // contiguous 128x128 tile
  const bf16* kbase = Kp + (size_t)(b*KVH_ + kv)*S_*DH_;
  const bf16* vbase = Vt + (size_t)(b*KVH_ + kv)*DH_*S_;

  // ---- kick off async staging of K/V tile 0 ----
  {
    const bf16* gk = kbase;                       // kb = 0: K tile contiguous 8KB
    unsigned lk = lds_off32(&bufK[0][0]);
#pragma unroll
    for (int u=0; u<2; ++u){
      int chunk = tid + u*256;                    // 512 16B chunks
      async_ld16(lk + chunk*16, gk + chunk*8);
    }
    unsigned lv = lds_off32(&bufV[0][0]);
#pragma unroll
    for (int u=0; u<2; ++u){
      int chunk = tid + u*256;
      int d = chunk >> 2, c4 = chunk & 3;         // V rows strided by S
      async_ld16(lv + chunk*16, vbase + (size_t)d*S_ + c4*8);
    }
  }

  // ---- TDM: one wave DMAs the whole Q tile (1-D, 16384 x 2B elements) ----
  if (tid < 32) {
    unsigned long long ga = (unsigned long long)(uintptr_t)qbase;
    v4u g0 = {0u,0u,0u,0u};
    g0.x = 1u;                                            // count = 1 (valid user D#)
    g0.y = lds_off32(&bufQ[0]);                           // lds_addr
    g0.z = (unsigned)ga;                                  // global_addr[31:0]
    g0.w = (unsigned)((ga >> 32) & 0x01FFFFFFu) | (2u << 30);   // addr[56:32] | type=2
    v8i g1 = {0,0,0,0,0,0,0,0};
    g1[0] = 0x00010000;                 // workgroup_mask=0, data_size=1 (2 bytes)
    g1[1] = (int)(0x4000u << 16);       // tensor_dim0 = 16384 (low16 -> bits 63:48)
    g1[2] = (int)(1u << 16);            // tensor_dim0 hi=0; tensor_dim1 = 1 (bits 95:80)
    g1[3] = (int)(0x4000u << 16);       // tile_dim0 = 16384 (bits 127:112)
    g1[4] = 1;                          // tile_dim1 = 1
    g1[5] = 16384;                      // tensor_dim0_stride low32
    g1[6] = (int)(0x4000u << 16);       // tensor_dim1_stride low16 (bits 223:208)
    v4i gz = {0,0,0,0};
    v8i gz8 = {0,0,0,0,0,0,0,0};
    __builtin_amdgcn_tensor_load_to_lds(g0, g1, gz, gz, gz8, 0);   // 6-arg form (clang-23)
    __builtin_amdgcn_s_wait_tensorcnt(0);
  }
  __syncthreads();                       // Q tile visible to all waves

  // preload Q fragments from LDS for the wave's 16 rows (all of DH=128)
  const bf16* qrow = &bufQ[(w*16 + c)*DH_];
  v16bf qf[4];
#pragma unroll
  for (int j=0;j<4;++j)
    qf[j] = load_frag32(qrow + 32*j + 16*half);

  v8f oacc[8];
#pragma unroll
  for (int t=0;t<8;++t) oacc[t] = zero8();
  float mrun[8], lrun[8];
#pragma unroll
  for (int r=0;r<8;++r){ mrun[r] = -1e30f; lrun[r] = 0.f; }

  const float scale = 0.088388347648318447f;   // 1/sqrt(128)
  const int ntiles = (q0 + 128) >> 5;          // causal: keys < q0+128

  for (int i=0; i<ntiles; ++i){
    const int kb = i*32;
    if (i+1 < ntiles){
      // stage next K/V tile into the other buffer
      const int nb = (i+1) & 1;
      const bf16* gk = kbase + (size_t)(kb+32)*DH_;
      unsigned lk = lds_off32(&bufK[nb][0]);
#pragma unroll
      for (int u=0; u<2; ++u){
        int chunk = tid + u*256;
        async_ld16(lk + chunk*16, gk + chunk*8);
      }
      unsigned lv = lds_off32(&bufV[nb][0]);
#pragma unroll
      for (int u=0; u<2; ++u){
        int chunk = tid + u*256;
        int d = chunk >> 2, c4 = chunk & 3;
        async_ld16(lv + chunk*16, vbase + (size_t)d*S_ + kb + 32 + c4*8);
      }
      asm volatile("s_wait_asynccnt 4" ::: "memory");   // tile i complete; 4 in flight
    } else {
      asm volatile("s_wait_asynccnt 0" ::: "memory");
    }
    __syncthreads();                                    // all waves' copies visible

    const bf16* Kt = &bufK[i&1][0];
    const bf16* Vb = &bufV[i&1][0];

    // ---- scores S = Q Kᵀ  (16 x 32 per wave), B-frags from LDS ----
    v8f sacc[2]; sacc[0] = zero8(); sacc[1] = zero8();
#pragma unroll
    for (int j=0;j<4;++j){
#pragma unroll
      for (int t=0;t<2;++t){
        v16bf bv = load_frag32(Kt + (16*t + c)*DH_ + 32*j + 16*half);
        sacc[t] = WMMA_BF16(qf[j], bv, sacc[t]);
      }
    }
    // ---- scale + analytic causal mask ----
#pragma unroll
    for (int t=0;t<2;++t){
#pragma unroll
      for (int r=0;r<8;++r){
        const int n  = kb + 16*t + c;
        const int sq = q0 + w*16 + rbase + r;
        const float sv = sacc[t][r]*scale;
        sacc[t][r] = (n > sq) ? -1e30f : sv;
      }
    }
    // ---- online softmax (row reductions across the 16 lanes of each half) ----
#pragma unroll
    for (int r=0;r<8;++r){
      float mx = fmaxf(sacc[0][r], sacc[1][r]);
#pragma unroll
      for (int m=1;m<16;m<<=1) mx = fmaxf(mx, __shfl_xor(mx, m, 32));
      const float mnew  = fmaxf(mrun[r], mx);
      const float alpha = __expf(mrun[r] - mnew);
      const float p0 = __expf(sacc[0][r] - mnew);
      const float p1 = __expf(sacc[1][r] - mnew);
      float ps = p0 + p1;
#pragma unroll
      for (int m=1;m<16;m<<=1) ps += __shfl_xor(ps, m, 32);
      lrun[r] = lrun[r]*alpha + ps;
      mrun[r] = mnew;
#pragma unroll
      for (int t=0;t<8;++t) oacc[t][r] *= alpha;
      const int mrow = rbase + r;
      ldsP[w][mrow][c]      = (bf16)p0;
      ldsP[w][mrow][16 + c] = (bf16)p1;
    }
    // ---- C-layout -> A-layout transpose through LDS, then O += P V (V from LDS) ----
    v16bf pa = load_frag32(&ldsP[w][c][16*half]);
#pragma unroll
    for (int t=0;t<8;++t){
      v16bf vv = load_frag32(Vb + (16*t + c)*32 + 16*half);
      oacc[t] = WMMA_BF16(pa, vv, oacc[t]);
    }
    __syncthreads();      // everyone done with buffer i&1 before it is re-staged
  }

  // ---- finalize: O / l, store to attn_out [B][S][H*DH] ----
#pragma unroll
  for (int t=0;t<8;++t){
#pragma unroll
    for (int r=0;r<8;++r){
      const int s = q0 + w*16 + rbase + r;
      const int d = 16*t + c;
      AO[((size_t)(b*S_ + s))*D_ + h*DH_ + d] = (bf16)(oacc[t][r] / lrun[r]);
    }
  }
}

// ------------------------------------------------------------------
// Output projection: out[B*S, 2048] = AO[B*S, 2048] @ Wo, f32 result.
// grid = (B*S/128, D/128), block = 256.
__global__ void __launch_bounds__(256) out_proj_kernel(
    const bf16* __restrict__ AO, const bf16* __restrict__ WoT, float* __restrict__ out)
{
  const int lane = threadIdx.x & 31;
  const int w    = threadIdx.x >> 5;
  const int c    = lane & 15;
  const int half = lane >> 4;
  const int khalf= half*8;
  const int r0   = blockIdx.x*128 + w*16;
  const int n0   = blockIdx.y*128;

  const bf16* arow = AO + (size_t)(r0 + c)*D_;
  v8f acc[8];
#pragma unroll
  for (int t=0;t<8;++t) acc[t] = zero8();

  for (int k0=0;k0<D_;k0+=32){
    v16bf a = load_frag32(arow + k0 + 16*half);
    __builtin_prefetch(arow + k0 + 128, 0, 0);
#pragma unroll
    for (int t=0;t<8;++t){
      const bf16* bp = WoT + (size_t)(n0 + 16*t + c)*D_ + k0 + 16*half;
      acc[t] = WMMA_BF16(a, load_frag32(bp), acc[t]);
    }
  }
  const int rbase = half*8;
#pragma unroll
  for (int t=0;t<8;++t)
#pragma unroll
    for (int r=0;r<8;++r)
      out[(size_t)(r0 + rbase + r)*D_ + n0 + 16*t + c] = acc[t][r];
}

// ------------------------------------------------------------------
extern "C" void kernel_launch(void* const* d_in, const int* in_sizes, int n_in,
                              void* d_out, int out_size, void* d_ws, size_t ws_size,
                              hipStream_t stream)
{
  (void)in_sizes; (void)n_in; (void)out_size; (void)ws_size;
  const float* x  = (const float*)d_in[0];
  // d_in[1] = mask (causal, computed analytically), d_in[2] = position_ids (arange) — unused
  const float* Wq = (const float*)d_in[3];
  const float* Wk = (const float*)d_in[4];
  const float* Wv = (const float*)d_in[5];
  const float* Wo = (const float*)d_in[6];
  float* out = (float*)d_out;

  bf16* p   = (bf16*)d_ws;
  bf16* xb  = p; p += (size_t)B_*S_*D_;        // 16 MB
  bf16* WqT = p; p += (size_t)D_*D_;           //  8 MB
  bf16* WkT = p; p += (size_t)(KVH_*DH_)*D_;   //  2 MB
  bf16* WvT = p; p += (size_t)(KVH_*DH_)*D_;   //  2 MB
  bf16* WoT = p; p += (size_t)D_*D_;           //  8 MB
  bf16* Qp  = p; p += (size_t)B_*H_*S_*DH_;    // 16 MB
  bf16* Kp  = p; p += (size_t)B_*KVH_*S_*DH_;  //  4 MB
  bf16* Vt  = p; p += (size_t)B_*KVH_*DH_*S_;  //  4 MB
  bf16* AO  = p; p += (size_t)B_*S_*D_;        // 16 MB   (total ~76 MB)

  const int nX = B_*S_*D_;
  cvt_f32_to_bf16_kernel<<<(nX+255)/256, 256, 0, stream>>>(x, xb, nX);
  transpose_to_bf16_kernel<<<(D_*D_      +255)/256, 256, 0, stream>>>(Wq, WqT, D_, D_);
  transpose_to_bf16_kernel<<<(D_*KVH_*DH_+255)/256, 256, 0, stream>>>(Wk, WkT, D_, KVH_*DH_);
  transpose_to_bf16_kernel<<<(D_*KVH_*DH_+255)/256, 256, 0, stream>>>(Wv, WvT, D_, KVH_*DH_);
  transpose_to_bf16_kernel<<<(D_*D_      +255)/256, 256, 0, stream>>>(Wo, WoT, D_, D_);

  proj_rope_kernel<<<dim3(S_/128, H_ + 2*KVH_, B_), 256, 0, stream>>>(xb, WqT, WkT, WvT, Qp, Kp, Vt);
  flash_attn_kernel<<<dim3(S_/128, H_, B_), 256, 0, stream>>>(Qp, Kp, Vt, AO);
  out_proj_kernel<<<dim3((B_*S_)/128, D_/128), 256, 0, stream>>>(AO, WoT, out);
}
